// ExtractorHead_58548994179388
// MI455X (gfx1250) — matrix operand order repaired and checked by reading, not compiled, not run on hardware
//
#include <hip/hip_runtime.h>
#include <math.h>

// ---------------- CDNA5 WMMA helpers (wave32, 16x16x32 bf16, f32 accum) ----

typedef __attribute__((ext_vector_type(16))) __bf16 v16bf;
typedef __attribute__((ext_vector_type(8)))  float  v8f;

#define HD    128
#define HD3   384
#define NRBF  20
#define PI_F  3.14159265358979323846f
#define CUT_F 6.0f

static __device__ __forceinline__ v8f wmma_bf16(v16bf a, v16bf b, v8f c) {
  // D = A(16x32) * B(32x16) + C(16x16), f32 accumulate
  return __builtin_amdgcn_wmma_f32_16x16x32_bf16(false, a, false, b, (short)0, c,
                                                 false, false);
}

// A-matrix 16x32 bf16 fragment from row-major fp32 source (on-the-fly cvt).
// ISA 7.12.2: lane L holds row M=L&15; lanes<16 K={0..7,16..23}, lanes>=16 K={8..15,24..31}
static __device__ __forceinline__ v16bf load_a_f32(const float* __restrict__ A, int ld,
                                                   int row0, int k0, int maxrow) {
  int lane = threadIdx.x & 31;
  int r = row0 + (lane & 15); if (r > maxrow) r = maxrow;   // clamp (no divergence)
  int hi = (lane & 16) ? 8 : 0;
  const float* p = A + (size_t)r * ld + k0;
  v16bf a;
#pragma unroll
  for (int t = 0; t < 8; ++t) {
    a[t]     = (__bf16)p[hi + t];
    a[8 + t] = (__bf16)p[16 + hi + t];
  }
  return a;
}

// A fragment from an LDS bf16 tile (16 x ld)
static __device__ __forceinline__ v16bf load_a_lds(const __bf16* __restrict__ A, int ld,
                                                   int k0) {
  int lane = threadIdx.x & 31;
  int r = lane & 15;
  int hi = (lane & 16) ? 8 : 0;
  const __bf16* p = A + r * ld + k0;
  v16bf a;
#pragma unroll
  for (int t = 0; t < 8; ++t) {
    a[t]     = p[hi + t];
    a[8 + t] = p[16 + hi + t];
  }
  return a;
}

// B-matrix 32x16 bf16 fragment from row-major fp32 weights W[K][ldb].
// lane holds column N=lane&15; lanes<16 K={0..7,16..23}, lanes>=16 K={8..15,24..31}
static __device__ __forceinline__ v16bf load_b_f32(const float* __restrict__ W, int ldb,
                                                   int k0, int col0) {
  int lane = threadIdx.x & 31;
  int n  = col0 + (lane & 15);
  int kb = k0 + ((lane & 16) ? 8 : 0);
  const float* p = W + (size_t)kb * ldb + n;
  v16bf b;
#pragma unroll
  for (int j = 0; j < 16; ++j) b[j] = (__bf16)p[(j < 8 ? j : j + 8) * ldb];
  return b;
}

// C-tile store with wave-uniform full/partial fast path
static __device__ __forceinline__ void store_c(float* __restrict__ Out, int ldc, v8f acc,
                                               float bias, int row0, int n, int mb,
                                               int M, bool full) {
  float* p = Out + (size_t)(row0 + mb) * ldc + n;
  if (full) {
#pragma unroll
    for (int rr = 0; rr < 8; ++rr) p[(size_t)rr * ldc] = acc[rr] + bias;
  } else {
#pragma unroll
    for (int rr = 0; rr < 8; ++rr)
      if (row0 + mb + rr < M) p[(size_t)rr * ldc] = acc[rr] + bias;
  }
}

static __device__ __forceinline__ float silu_f(float x) {
  return x / (1.0f + __expf(-x));
}

// ---------------- Kernels ------------------------------------------------

// Edge geometry: unit vector, fcut, sinc RBF, and edge_dist output
__global__ void k_geom(const float* __restrict__ pos, const int* __restrict__ row,
                       const int* __restrict__ col, const float* __restrict__ cofs,
                       const float* __restrict__ cell, float* __restrict__ unitv,
                       float* __restrict__ fcut, float* __restrict__ rbf,
                       float* __restrict__ dist_out, int E) {
  int e = blockIdx.x * blockDim.x + threadIdx.x;
  if (e >= E) return;
  int r = row[e], c = col[e];
  float c0 = cofs[e * 3 + 0], c1 = cofs[e * 3 + 1], c2 = cofs[e * 3 + 2];
  float d3[3];
#pragma unroll
  for (int d = 0; d < 3; ++d) {
    float off = c0 * cell[0 * 3 + d] + c1 * cell[1 * 3 + d] + c2 * cell[2 * 3 + d];
    d3[d] = pos[r * 3 + d] - pos[c * 3 + d] + off;
  }
  float dist = sqrtf(d3[0] * d3[0] + d3[1] * d3[1] + d3[2] * d3[2]);
  dist_out[e] = dist;
  float inv = 1.0f / dist;
  unitv[e * 3 + 0] = d3[0] * inv;
  unitv[e * 3 + 1] = d3[1] * inv;
  unitv[e * 3 + 2] = d3[2] * inv;
  fcut[e] = (dist < CUT_F) ? 0.5f * (__cosf(PI_F * dist / CUT_F) + 1.0f) : 0.0f;
  float w = dist * PI_F / CUT_F;
#pragma unroll
  for (int k = 1; k <= NRBF; ++k) rbf[(size_t)e * NRBF + (k - 1)] = __sinf(w * k) * inv;
}

// ns = embed[z], nv = 0
__global__ void k_init(const int* __restrict__ z, const float* __restrict__ embed,
                       float* __restrict__ ns, float* __restrict__ nv, int N) {
  int idx = blockIdx.x * blockDim.x + threadIdx.x;
  if (idx >= N * HD3) return;
  nv[idx] = 0.0f;
  if (idx < N * HD) {
    int n = idx >> 7, h = idx & 127;
    ns[idx] = embed[(size_t)z[n] * HD + h];
  }
}

// Fused Out = silu(A@W1+b1)@W2+b2 : A is MxK1 fp32; hidden 128 in LDS (bf16);
// one wave owns a 16-row tile. All GEMMs via v_wmma_f32_16x16x32_bf16.
template <int K1, int DOUT>
__global__ __launch_bounds__(128, 1) void k_mlp2(
    const float* __restrict__ A, int M, const float* __restrict__ W1,
    const float* __restrict__ b1, const float* __restrict__ W2,
    const float* __restrict__ b2, float* __restrict__ Out) {
  __shared__ __bf16 hid[4 * 16 * HD];
  int wave = threadIdx.x >> 5;
  int tile = blockIdx.x * 4 + wave;
  int row0 = tile * 16;
  if (row0 >= M) return;                      // wave-uniform exit, EXEC stays full
  bool full = (row0 + 16 <= M);               // wave-uniform
  int maxrow = M - 1;
  int lane = threadIdx.x & 31;
  int nc   = (lane & 15);
  int mb   = (lane & 16) ? 8 : 0;

  v16bf afr[K1 / 32];
#pragma unroll
  for (int kc = 0; kc < K1 / 32; ++kc) afr[kc] = load_a_f32(A, K1, row0, kc * 32, maxrow);

  __bf16* hw = hid + wave * (16 * HD);
  // phase 1: hidden = silu(A@W1+b1)  -> LDS bf16 tile
#pragma unroll
  for (int ct = 0; ct < HD / 16; ++ct) {
    v8f acc = {};
#pragma unroll
    for (int kc = 0; kc < K1 / 32; ++kc)
      acc = wmma_bf16(afr[kc], load_b_f32(W1, HD, kc * 32, ct * 16), acc);
    int n = ct * 16 + nc;
    float bias = b1[n];
#pragma unroll
    for (int rr = 0; rr < 8; ++rr) {
      float x = acc[rr] + bias;
      hw[(mb + rr) * HD + n] = (__bf16)silu_f(x);
    }
  }
  // phase 2: Out = hidden@W2 + b2
  v16bf hfr[4];
#pragma unroll
  for (int kc = 0; kc < 4; ++kc) hfr[kc] = load_a_lds(hw, HD, kc * 32);
#pragma unroll
  for (int ct = 0; ct < DOUT / 16; ++ct) {
    v8f acc = {};
#pragma unroll
    for (int kc = 0; kc < 4; ++kc)
      acc = wmma_bf16(hfr[kc], load_b_f32(W2, DOUT, kc * 32, ct * 16), acc);
    int n = ct * 16 + nc;
    store_c(Out, DOUT, acc, b2[n], row0, n, mb, M, full);
  }
}

// Uv = nv@uw+ub ; Vv = nv@vw+vb  over M=3N rows, K=128, Dout=128 (shared A frags)
__global__ __launch_bounds__(128, 1) void k_uv(
    const float* __restrict__ nv, int M, const float* __restrict__ uw,
    const float* __restrict__ ub, const float* __restrict__ vw,
    const float* __restrict__ vb, float* __restrict__ Uv, float* __restrict__ Vv) {
  int tile = blockIdx.x * 4 + (threadIdx.x >> 5);
  int row0 = tile * 16;
  if (row0 >= M) return;
  bool full = (row0 + 16 <= M);
  int maxrow = M - 1;
  int lane = threadIdx.x & 31;
  int nc = (lane & 15), mb = (lane & 16) ? 8 : 0;

  v16bf afr[4];
#pragma unroll
  for (int kc = 0; kc < 4; ++kc) afr[kc] = load_a_f32(nv, HD, row0, kc * 32, maxrow);

#pragma unroll
  for (int which = 0; which < 2; ++which) {
    const float* W = which ? vw : uw;
    const float* B = which ? vb : ub;
    float* O = which ? Vv : Uv;
#pragma unroll
    for (int ct = 0; ct < HD / 16; ++ct) {
      v8f acc = {};
#pragma unroll
      for (int kc = 0; kc < 4; ++kc)
        acc = wmma_bf16(afr[kc], load_b_f32(W, HD, kc * 32, ct * 16), acc);
      int n = ct * 16 + nc;
      store_c(O, HD, acc, B[n], row0, n, mb, M, full);
    }
  }
}

// Per-edge message + scatter. One wave per edge. filt_w slice (20x384) in LDS.
__global__ __launch_bounds__(256) void k_edge(
    const int* __restrict__ row, const int* __restrict__ col,
    const float* __restrict__ rbf, const float* __restrict__ fcut,
    const float* __restrict__ unitv, const float* __restrict__ sout,
    const float* __restrict__ nv_prev, const float* __restrict__ fw,
    const float* __restrict__ fb, float* __restrict__ ns, float* __restrict__ nv,
    int E) {
  __shared__ float Wl[NRBF * HD3];
  __shared__ float Bl[HD3];
  for (int i = threadIdx.x; i < NRBF * HD3; i += blockDim.x) Wl[i] = fw[i];
  for (int i = threadIdx.x; i < HD3; i += blockDim.x) Bl[i] = fb[i];
  __syncthreads();

  int e = blockIdx.x * 8 + (threadIdx.x >> 5);
  if (e >= E) return;
  int lane = threadIdx.x & 31;
  int r = row[e], c = col[e];

  float rb[NRBF];
#pragma unroll
  for (int k = 0; k < NRBF; ++k) rb[k] = rbf[(size_t)e * NRBF + k];
  float fc = fcut[e];
  float u0 = unitv[e * 3 + 0], u1 = unitv[e * 3 + 1], u2 = unitv[e * 3 + 2];

  const float* sc  = sout + (size_t)c * HD3;
  const float* nc_ = nv_prev + (size_t)c * HD3;
  float* nsr = ns + (size_t)r * HD;
  float* nvr = nv + (size_t)r * HD3;

#pragma unroll
  for (int q = 0; q < 4; ++q) {
    int h = lane + 32 * q;
    float fv = Bl[h], fe = Bl[HD + h], fs = Bl[2 * HD + h];
#pragma unroll
    for (int k = 0; k < NRBF; ++k) {
      float rk = rb[k];
      fv = fmaf(rk, Wl[k * HD3 + h], fv);
      fe = fmaf(rk, Wl[k * HD3 + HD + h], fe);
      fs = fmaf(rk, Wl[k * HD3 + 2 * HD + h], fs);
    }
    float gv = fv * fc * sc[h];
    float ge = fe * fc * sc[HD + h];
    float ms = fs * fc * sc[2 * HD + h];
    atomicAdd(nsr + h, ms);
    atomicAdd(nvr + h,          fmaf(nc_[h],          gv, ge * u0));
    atomicAdd(nvr + HD + h,     fmaf(nc_[HD + h],     gv, ge * u1));
    atomicAdd(nvr + 2 * HD + h, fmaf(nc_[2 * HD + h], gv, ge * u2));
  }
}

// mlp_in = [ ||Vv||_dim , ns ]  (N x 256 fp32)
__global__ void k_prep(const float* __restrict__ Vv, const float* __restrict__ ns,
                       float* __restrict__ mlp_in, int N) {
  int idx = blockIdx.x * blockDim.x + threadIdx.x;
  if (idx >= N * HD) return;
  int n = idx >> 7, h = idx & 127;
  float v0 = Vv[(size_t)n * HD3 + h];
  float v1 = Vv[(size_t)n * HD3 + HD + h];
  float v2 = Vv[(size_t)n * HD3 + 2 * HD + h];
  mlp_in[(size_t)n * 256 + h] = sqrtf(v0 * v0 + v1 * v1 + v2 * v2);
  mlp_in[(size_t)n * 256 + HD + h] = ns[idx];
}

// apply update: inner = sum_d Uv*Vv ; nv += a_vv*Uv ; ns += a_sv*inner + a_ss
__global__ void k_apply(const float* __restrict__ m, const float* __restrict__ Uv,
                        const float* __restrict__ Vv, float* __restrict__ nv,
                        float* __restrict__ ns, int N) {
  int idx = blockIdx.x * blockDim.x + threadIdx.x;
  if (idx >= N * HD) return;
  int n = idx >> 7, h = idx & 127;
  float a_vv = m[(size_t)n * HD3 + h];
  float a_sv = m[(size_t)n * HD3 + HD + h];
  float a_ss = m[(size_t)n * HD3 + 2 * HD + h];
  float inner = 0.0f;
#pragma unroll
  for (int d = 0; d < 3; ++d) {
    size_t o = (size_t)n * HD3 + d * HD + h;
    float u = Uv[o], v = Vv[o];
    inner = fmaf(u, v, inner);
    nv[o] += a_vv * u;
  }
  ns[idx] += fmaf(a_sv, inner, a_ss);
}

// ---------------- Host launch --------------------------------------------

extern "C" void kernel_launch(void* const* d_in, const int* in_sizes, int n_in,
                              void* d_out, int out_size, void* d_ws, size_t ws_size,
                              hipStream_t stream) {
  const int*   z     = (const int*)d_in[0];
  const float* pos   = (const float*)d_in[1];
  const int*   ei    = (const int*)d_in[2];
  const float* cofs  = (const float*)d_in[3];
  const float* cell  = (const float*)d_in[4];
  const float* embed = (const float*)d_in[5];
  const float* msg_w1 = (const float*)d_in[6];
  const float* msg_b1 = (const float*)d_in[7];
  const float* msg_w2 = (const float*)d_in[8];
  const float* msg_b2 = (const float*)d_in[9];
  const float* filt_w = (const float*)d_in[10];
  const float* filt_b = (const float*)d_in[11];
  const float* upd_uw = (const float*)d_in[12];
  const float* upd_ub = (const float*)d_in[13];
  const float* upd_vw = (const float*)d_in[14];
  const float* upd_vb = (const float*)d_in[15];
  const float* upd_mw1 = (const float*)d_in[16];
  const float* upd_mb1 = (const float*)d_in[17];
  const float* upd_mw2 = (const float*)d_in[18];
  const float* upd_mb2 = (const float*)d_in[19];
  const float* head_w1 = (const float*)d_in[20];
  const float* head_b1 = (const float*)d_in[21];
  const float* head_w2 = (const float*)d_in[22];
  const float* head_b2 = (const float*)d_in[23];

  const int N = in_sizes[0];
  const int E = in_sizes[2] / 2;
  const int* row = ei;
  const int* col = ei + E;

  float* out_v    = (float*)d_out;
  float* dist_out = out_v + (size_t)N * HD;

  // workspace carve (fp32)
  float* ws = (float*)d_ws;
  size_t o = 0;
  float* ns      = ws + o; o += (size_t)N * HD;
  float* nv      = ws + o; o += (size_t)N * HD3;
  float* nv_prev = ws + o; o += (size_t)N * HD3;
  float* buf384  = ws + o; o += (size_t)N * HD3;   // sout, then update-mlp out
  float* Uv      = ws + o; o += (size_t)N * HD3;
  float* Vv      = ws + o; o += (size_t)N * HD3;
  float* mlp_in  = ws + o; o += (size_t)N * 256;
  float* unitv   = ws + o; o += (size_t)E * 3;
  float* fcut    = ws + o; o += (size_t)E;
  float* rbf     = ws + o; o += (size_t)E * NRBF;
  (void)ws_size; (void)n_in; (void)out_size;

  k_geom<<<(E + 255) / 256, 256, 0, stream>>>(pos, row, col, cofs, cell, unitv, fcut,
                                              rbf, dist_out, E);
  k_init<<<(N * HD3 + 255) / 256, 256, 0, stream>>>(z, embed, ns, nv, N);

  const int ntiles  = (N + 15) / 16;
  const int nblk    = (ntiles + 3) / 4;
  const int m3      = 3 * N;
  const int t3      = (m3 + 15) / 16;
  const int blk3    = (t3 + 3) / 4;

  for (int i = 0; i < 3; ++i) {
    // sout = silu(ns@msg_w1+b1)@msg_w2+b2   (N x 384)
    k_mlp2<HD, HD3><<<nblk, 128, 0, stream>>>(
        ns, N, msg_w1 + (size_t)i * HD * HD, msg_b1 + (size_t)i * HD,
        msg_w2 + (size_t)i * HD * HD3, msg_b2 + (size_t)i * HD3, buf384);
    // snapshot nv for gather-side reads
    hipMemcpyAsync(nv_prev, nv, (size_t)N * HD3 * sizeof(float),
                   hipMemcpyDeviceToDevice, stream);
    // edge messages + scatter
    k_edge<<<(E + 7) / 8, 256, 0, stream>>>(
        row, col, rbf, fcut, unitv, buf384, nv_prev,
        filt_w + (size_t)i * NRBF * HD3, filt_b + (size_t)i * HD3, ns, nv, E);
    // Uv, Vv
    k_uv<<<blk3, 128, 0, stream>>>(nv, m3, upd_uw + (size_t)i * HD * HD,
                                   upd_ub + (size_t)i * HD,
                                   upd_vw + (size_t)i * HD * HD,
                                   upd_vb + (size_t)i * HD, Uv, Vv);
    // mlp_in = [||Vv||, ns]
    k_prep<<<(N * HD + 255) / 256, 256, 0, stream>>>(Vv, ns, mlp_in, N);
    // m = silu(mlp_in@mw1+mb1)@mw2+mb2   (N x 384)
    k_mlp2<256, HD3><<<nblk, 128, 0, stream>>>(
        mlp_in, N, upd_mw1 + (size_t)i * 256 * HD, upd_mb1 + (size_t)i * HD,
        upd_mw2 + (size_t)i * HD * HD3, upd_mb2 + (size_t)i * HD3, buf384);
    // apply update
    k_apply<<<(N * HD + 255) / 256, 256, 0, stream>>>(buf384, Uv, Vv, nv, ns, N);
  }

  // head: v = silu(ns@hw1+hb1)@hw2+hb2 -> d_out
  k_mlp2<HD, HD><<<nblk, 128, 0, stream>>>(ns, N, head_w1, head_b1, head_w2, head_b2,
                                           out_v);
}
